// MultiHeadAttention_721554505960
// MI455X (gfx1250) — compile-verified, matrix operand order checked by reading, not run on hardware
//
#include <hip/hip_runtime.h>

// ---------------------------------------------------------------------------
// MI455X / gfx1250 multi-head causal attention (fwd), compile-only target.
// Shapes: B=1, S=4096, DIM=512, H=8, d=64, causal.
// Precision: f16 multiplicands, f32 accumulation (V_WMMA_F32_16X16X32_F16).
//
// Stage 0: cvt      : fp32 -> f16 copies of q/k/v and all weights (streaming)
// Stage 1: qkv_proj : f16 [S,512] x W^T -> f16 Qh/Kh [S,512], Vt [512,S]
// Stage 2: attn     : flash attention; 4 waves/block share K/V tiles staged
//                     into LDS by the Tensor Data Mover (double-buffered,
//                     TENSORcnt-synchronized).
// Stage 3: out_proj : f16 [S,512] x Wf^T + bf -> fp32 [S,512]
// Workspace: Qh+Kh+Vt+O (16MB) + Xq/Xk/Xv f16 (12MB) + weights f16 (2MB).
// ---------------------------------------------------------------------------

typedef __attribute__((ext_vector_type(16))) _Float16 v16h;
typedef __attribute__((ext_vector_type(8)))  _Float16 v8h;
typedef __attribute__((ext_vector_type(8)))  float    v8f;
typedef __attribute__((ext_vector_type(4)))  unsigned u32x4;
typedef __attribute__((ext_vector_type(8)))  unsigned u32x8;

#define S_SEQ 4096
#define DIM   512
#define NHEAD 8
#define DK    64

__device__ __forceinline__ v8f wmma_f16(v16h a, v16h b, v8f c) {
  return __builtin_amdgcn_wmma_f32_16x16x32_f16(
      false, a, false, b, (short)0, c, false, false);
}

// A/B fragment loader, row-major f16 source (ISA 7.12.2, 16-bit A 16x32):
// lane = {row = lane&15, half = lane>>4}; element i holds
//   k = 8*half + i (i<8) ; k = 16 + 8*half + (i-8) (i>=8)
__device__ __forceinline__ v16h load_frag_h(const _Float16* rowp, int half) {
  v8h c0 = *reinterpret_cast<const v8h*>(rowp + 8 * half);
  v8h c1 = *reinterpret_cast<const v8h*>(rowp + 16 + 8 * half);
  v16h f;
#pragma unroll
  for (int i = 0; i < 8; ++i) { f[i] = c0[i]; f[i + 8] = c1[i]; }
  return f;
}

// ---------------------------------------------------------------------------
// Stage 0: streaming fp32 -> f16 conversion (n must be a multiple of 8).
// ---------------------------------------------------------------------------
__global__ __launch_bounds__(256) void f32_to_f16(
    const float* __restrict__ src, _Float16* __restrict__ dst, int n) {
  const int i = (blockIdx.x * 256 + threadIdx.x) * 8;
  if (i < n) {
    v8f x = *reinterpret_cast<const v8f*>(src + i);
    v8h y;
#pragma unroll
    for (int j = 0; j < 8; ++j) y[j] = (_Float16)x[j];
    *reinterpret_cast<v8h*>(dst + i) = y;
  }
}

// ---------------------------------------------------------------------------
// Tensor Data Mover: 2D f16 tile load, Global -> LDS (D# per
// CDNA5 ISA; 2-group form => up-to-2D tensors).
// ---------------------------------------------------------------------------
__device__ __forceinline__ void tdm_load_2d_f16(
    const _Float16* lds_dst, const _Float16* gsrc,
    unsigned tensor_d0, unsigned tensor_d1,
    unsigned tile_d0, unsigned tile_d1, unsigned stride0) {
  unsigned long long ga = (unsigned long long)(const void*)gsrc;
  unsigned lds_off = (unsigned)(unsigned long long)(const void*)lds_dst;
  u32x4 g0;
  g0[0] = 1u;                                       // count=1, user D#
  g0[1] = lds_off;                                  // lds_addr
  g0[2] = (unsigned)(ga & 0xFFFFFFFFu);             // global_addr[31:0]
  g0[3] = (unsigned)((ga >> 32) & 0x01FFFFFFu)      // global_addr[56:32]
          | (2u << 30);                             // type=2 ("image")
  u32x8 g1;
  g1[0] = (1u << 16);                               // data_size=1 (2B), mask=0
  g1[1] = (tensor_d0 & 0xFFFFu) << 16;              // tensor_dim0[15:0]
  g1[2] = ((tensor_d0 >> 16) & 0xFFFFu)             // tensor_dim0[31:16]
          | ((tensor_d1 & 0xFFFFu) << 16);          // tensor_dim1[15:0]
  g1[3] = ((tensor_d1 >> 16) & 0xFFFFu)             // tensor_dim1[31:16]
          | ((tile_d0 & 0xFFFFu) << 16);            // tile_dim0
  g1[4] = (tile_d1 & 0xFFFFu);                      // tile_dim1, tile_dim2=0
  g1[5] = stride0;                                  // tensor_dim0_stride[31:0]
  g1[6] = 0;                                        // stride0[47:32], stride1
  g1[7] = 0;
  asm volatile("tensor_load_to_lds %0, %1" :: "s"(g0), "s"(g1) : "memory");
}

// ---------------------------------------------------------------------------
// Stage 1: QKV projections, all-f16. grid = (S/16, DIM/64, 3), block = 32.
// out[s][o] = sum_i X[s][i] * W[o][i]; W rows contiguous == B-fragment shape.
// ---------------------------------------------------------------------------
__global__ __launch_bounds__(32) void qkv_proj(
    const _Float16* __restrict__ xq, const _Float16* __restrict__ xk,
    const _Float16* __restrict__ xv, const _Float16* __restrict__ wq,
    const _Float16* __restrict__ wk, const _Float16* __restrict__ wv,
    _Float16* __restrict__ Qh, _Float16* __restrict__ Kh,
    _Float16* __restrict__ Vt) {
  const int which = blockIdx.z;
  const _Float16* X = (which == 0) ? xq : (which == 1) ? xk : xv;
  const _Float16* W = (which == 0) ? wq : (which == 1) ? wk : wv;

  const int i0   = blockIdx.x * 16;
  const int n0   = blockIdx.y * 64;
  const int lane = threadIdx.x & 31;
  const int half = lane >> 4;
  const int lr   = lane & 15;

  v8f acc[4] = {};
  const _Float16* arow = X + (size_t)(i0 + lr) * DIM;
  const _Float16* brow[4];
#pragma unroll
  for (int nt = 0; nt < 4; ++nt)
    brow[nt] = W + (size_t)(n0 + nt * 16 + lr) * DIM;

  for (int kk = 0; kk < DIM; kk += 32) {
    v16h a = load_frag_h(arow + kk, half);
    v16h b[4];
#pragma unroll
    for (int nt = 0; nt < 4; ++nt) b[nt] = load_frag_h(brow[nt] + kk, half);
#pragma unroll
    for (int nt = 0; nt < 4; ++nt) acc[nt] = wmma_f16(a, b[nt], acc[nt]);
  }

#pragma unroll
  for (int nt = 0; nt < 4; ++nt) {
    const int col = n0 + nt * 16 + lr;
#pragma unroll
    for (int r = 0; r < 8; ++r) {
      const int row = i0 + r + 8 * half;
      const _Float16 val = (_Float16)acc[nt][r];
      if (which == 2) {
        Vt[(size_t)col * S_SEQ + row] = val;        // V stored transposed
      } else {
        _Float16* O = (which == 0) ? Qh : Kh;
        O[(size_t)row * DIM + col] = val;
      }
    }
  }
}

// ---------------------------------------------------------------------------
// Stage 2: flash attention. grid = (S/64, NHEAD), block = 128 (4 waves).
// Block owns 64 query rows of one head; wave w owns rows [i0+16w, i0+16w+16).
// K/V 32-key tiles are TDM-staged into double-buffered LDS shared by all
// 4 waves; wave 0 drives the DMA and TENSORcnt waits.
// ---------------------------------------------------------------------------
__global__ __launch_bounds__(128) void attn(
    const _Float16* __restrict__ Qh, const _Float16* __restrict__ Kh,
    const _Float16* __restrict__ Vt, _Float16* __restrict__ O) {
  __shared__ _Float16 Kb[2][32 * 64];   // [buf][key][d]   8 KB
  __shared__ _Float16 Vb[2][64 * 32];   // [buf][d][key]   8 KB
  __shared__ _Float16 Plds[4][16 * 32]; // per-wave P tile 4 KB

  const int h    = blockIdx.y;
  const int i0b  = blockIdx.x * 64;
  const int wid  = threadIdx.x >> 5;
  const int lane = threadIdx.x & 31;
  const int half = lane >> 4;
  const int lr   = lane & 15;
  const int i0   = i0b + wid * 16;

  const _Float16* qrow = Qh + (size_t)(i0 + lr) * DIM + h * DK;
  const v16h aq0 = load_frag_h(qrow, half);
  const v16h aq1 = load_frag_h(qrow + 32, half);

  v8f   acc[4] = {};
  float run_m[8], run_l[8];
#pragma unroll
  for (int r = 0; r < 8; ++r) { run_m[r] = -1e30f; run_l[r] = 0.f; }

  const int nblocks = (i0b + 64) / 32;

  if (wid == 0) {                                 // prologue: stage block 0
    tdm_load_2d_f16(&Kb[0][0], Kh + h * DK, DIM, S_SEQ, DK, 32, DIM);
    tdm_load_2d_f16(&Vb[0][0], Vt + (size_t)(h * DK) * S_SEQ,
                    S_SEQ, DIM, 32, DK, S_SEQ);
  }

  for (int jb = 0; jb < nblocks; ++jb) {
    const int j0  = jb * 32;
    const int buf = jb & 1;

    if (wid == 0) {
      if (jb + 1 < nblocks) {                     // prefetch next tile pair
        const int jn = j0 + 32;
        tdm_load_2d_f16(&Kb[buf ^ 1][0], Kh + (size_t)jn * DIM + h * DK,
                        DIM, S_SEQ, DK, 32, DIM);
        tdm_load_2d_f16(&Vb[buf ^ 1][0], Vt + (size_t)(h * DK) * S_SEQ + jn,
                        S_SEQ, DIM, 32, DK, S_SEQ);
        __builtin_amdgcn_s_wait_tensorcnt(2);     // current pair landed
      } else {
        __builtin_amdgcn_s_wait_tensorcnt(0);
      }
    }
    __syncthreads();                              // publish buffer `buf`

    if (j0 <= i0 + 15) {                          // wave has visible keys
      const _Float16* kb = &Kb[buf][0];
      const _Float16* vb = &Vb[buf][0];

      // ---- scores: 16x32 block as two 16x16 C tiles ------------------------
      v16h bk[4];
      bk[0] = load_frag_h(kb + lr * 64, half);
      bk[1] = load_frag_h(kb + lr * 64 + 32, half);
      bk[2] = load_frag_h(kb + (16 + lr) * 64, half);
      bk[3] = load_frag_h(kb + (16 + lr) * 64 + 32, half);
      v8f s0 = {}, s1 = {};
      s0 = wmma_f16(aq0, bk[0], s0);
      s0 = wmma_f16(aq1, bk[1], s0);
      s1 = wmma_f16(aq0, bk[2], s1);
      s1 = wmma_f16(aq1, bk[3], s1);

      const bool maskblk = (j0 + 31 > i0);
      const float scl = 0.125f;                   // 1/sqrt(64)
#pragma unroll
      for (int r = 0; r < 8; ++r) {
        s0[r] *= scl; s1[r] *= scl;
        if (maskblk) {
          const int gi = i0 + r + 8 * half;
          if (j0 + lr > gi)      s0[r] = -1e30f;
          if (j0 + 16 + lr > gi) s1[r] = -1e30f;
        }
      }

      // ---- online softmax (row reductions inside 16-lane halves) -----------
      float bm[8];
#pragma unroll
      for (int r = 0; r < 8; ++r) bm[r] = fmaxf(s0[r], s1[r]);
#pragma unroll
      for (int off = 8; off >= 1; off >>= 1)
#pragma unroll
        for (int r = 0; r < 8; ++r)
          bm[r] = fmaxf(bm[r], __shfl_xor(bm[r], off, 32));

      float p0[8], p1[8], rs[8], alpha[8];
#pragma unroll
      for (int r = 0; r < 8; ++r) {
        const float nm = fmaxf(run_m[r], bm[r]);
        alpha[r] = __expf(run_m[r] - nm);
        p0[r]    = __expf(s0[r] - nm);
        p1[r]    = __expf(s1[r] - nm);
        run_m[r] = nm;
        rs[r]    = p0[r] + p1[r];
      }
#pragma unroll
      for (int off = 8; off >= 1; off >>= 1)
#pragma unroll
        for (int r = 0; r < 8; ++r) rs[r] += __shfl_xor(rs[r], off, 32);
#pragma unroll
      for (int r = 0; r < 8; ++r) run_l[r] = run_l[r] * alpha[r] + rs[r];
#pragma unroll
      for (int nt = 0; nt < 4; ++nt)
#pragma unroll
        for (int r = 0; r < 8; ++r) acc[nt][r] *= alpha[r];

      // ---- P: C-layout -> A-fragment layout via per-wave LDS tile ----------
      _Float16* pw = &Plds[wid][0];
#pragma unroll
      for (int r = 0; r < 8; ++r) {
        pw[(r + 8 * half) * 32 + lr]      = (_Float16)p0[r];
        pw[(r + 8 * half) * 32 + 16 + lr] = (_Float16)p1[r];
      }
      const v16h ap = load_frag_h(pw + lr * 32, half);

      // ---- PV: O += P(16x32) * V(32x64) ------------------------------------
      v16h bv[4];
#pragma unroll
      for (int nt = 0; nt < 4; ++nt)
        bv[nt] = load_frag_h(vb + (nt * 16 + lr) * 32, half);
#pragma unroll
      for (int nt = 0; nt < 4; ++nt) acc[nt] = wmma_f16(ap, bv[nt], acc[nt]);
    }

    __syncthreads();  // all reads of `buf` done before its next TDM overwrite
  }

#pragma unroll
  for (int r = 0; r < 8; ++r) run_l[r] = 1.f / run_l[r];
#pragma unroll
  for (int nt = 0; nt < 4; ++nt)
#pragma unroll
    for (int r = 0; r < 8; ++r)
      O[(size_t)(i0 + r + 8 * half) * DIM + h * DK + nt * 16 + lr] =
          (_Float16)(acc[nt][r] * run_l[r]);
}

// ---------------------------------------------------------------------------
// Stage 3: output projection + bias. grid = (S/16, DIM/64), block = 32.
// ---------------------------------------------------------------------------
__global__ __launch_bounds__(32) void out_proj(
    const _Float16* __restrict__ A, const _Float16* __restrict__ wf,
    const float* __restrict__ bf, float* __restrict__ out) {
  const int i0   = blockIdx.x * 16;
  const int n0   = blockIdx.y * 64;
  const int lane = threadIdx.x & 31;
  const int half = lane >> 4;
  const int lr   = lane & 15;

  v8f acc[4] = {};
  const _Float16* arow = A + (size_t)(i0 + lr) * DIM;
  const _Float16* brow[4];
#pragma unroll
  for (int nt = 0; nt < 4; ++nt)
    brow[nt] = wf + (size_t)(n0 + nt * 16 + lr) * DIM;

  for (int kk = 0; kk < DIM; kk += 32) {
    v16h a = load_frag_h(arow + kk, half);
    v16h b[4];
#pragma unroll
    for (int nt = 0; nt < 4; ++nt) b[nt] = load_frag_h(brow[nt] + kk, half);
#pragma unroll
    for (int nt = 0; nt < 4; ++nt) acc[nt] = wmma_f16(a, b[nt], acc[nt]);
  }

#pragma unroll
  for (int nt = 0; nt < 4; ++nt) {
    const int col  = n0 + nt * 16 + lr;
    const float bb = bf[col];
#pragma unroll
    for (int r = 0; r < 8; ++r)
      out[(size_t)(i0 + r + 8 * half) * DIM + col] = acc[nt][r] + bb;
  }
}

// ---------------------------------------------------------------------------
extern "C" void kernel_launch(void* const* d_in, const int* in_sizes, int n_in,
                              void* d_out, int out_size, void* d_ws,
                              size_t ws_size, hipStream_t stream) {
  (void)in_sizes; (void)n_in; (void)out_size; (void)ws_size;
  const float* q  = (const float*)d_in[0];
  const float* k  = (const float*)d_in[1];
  const float* v  = (const float*)d_in[2];
  const float* Wq = (const float*)d_in[3];
  const float* Wk = (const float*)d_in[4];
  const float* Wv = (const float*)d_in[5];
  const float* Wf = (const float*)d_in[6];
  const float* bf = (const float*)d_in[7];

  const size_t NX = (size_t)S_SEQ * DIM;   // 2M elems
  const size_t NW = (size_t)DIM * DIM;     // 256K elems

  _Float16* Qh  = (_Float16*)d_ws;         // [S][512]
  _Float16* Kh  = Qh + NX;                 // [S][512]
  _Float16* Vt  = Kh + NX;                 // [512][S]
  _Float16* O   = Vt + NX;                 // [S][512]
  _Float16* xq  = O + NX;                  // f16 copies of inputs
  _Float16* xk  = xq + NX;
  _Float16* xv  = xk + NX;
  _Float16* wqh = xv + NX;                 // f16 copies of weights
  _Float16* wkh = wqh + NW;
  _Float16* wvh = wkh + NW;
  _Float16* wfh = wvh + NW;
  float* out    = (float*)d_out;

  // Stage 0: streaming fp32 -> f16 (bandwidth-bound, ~45MB total).
  const int TB = 256;
  f32_to_f16<<<(int)(NX / 8 / TB), TB, 0, stream>>>(q, xq, (int)NX);
  f32_to_f16<<<(int)(NX / 8 / TB), TB, 0, stream>>>(k, xk, (int)NX);
  f32_to_f16<<<(int)(NX / 8 / TB), TB, 0, stream>>>(v, xv, (int)NX);
  f32_to_f16<<<(int)(NW / 8 / TB), TB, 0, stream>>>(Wq, wqh, (int)NW);
  f32_to_f16<<<(int)(NW / 8 / TB), TB, 0, stream>>>(Wk, wkh, (int)NW);
  f32_to_f16<<<(int)(NW / 8 / TB), TB, 0, stream>>>(Wv, wvh, (int)NW);
  f32_to_f16<<<(int)(NW / 8 / TB), TB, 0, stream>>>(Wf, wfh, (int)NW);

  dim3 g1(S_SEQ / 16, DIM / 64, 3);
  qkv_proj<<<g1, 32, 0, stream>>>(xq, xk, xv, wqh, wkh, wvh, Qh, Kh, Vt);

  dim3 g2(S_SEQ / 64, NHEAD);
  attn<<<g2, 128, 0, stream>>>(Qh, Kh, Vt, O);

  dim3 g3(S_SEQ / 16, DIM / 64);
  out_proj<<<g3, 32, 0, stream>>>(O, wfh, bf, out);
}
